// S5Block_18373870092481
// MI455X (gfx1250) — matrix operand order repaired and checked by reading, not compiled, not run on hardware
//
#include <hip/hip_runtime.h>
#include <hip/hip_bf16.h>
#include <math.h>

// ---------------------------------------------------------------------------
// S5 block for MI455X (gfx1250, wave32, WMMA 16x16x32 f16 -> f32 accum)
//
// Shapes (hardcoded from reference): B=8, T=2048, DIM=1024, H=16, D=S=64.
//
// Pipeline:
//   k0: RMSNorm -> xn (f16) in [H][B][T][D] layout (contiguous per head/step)
//   k1: w_out -> f16 (kept [n][k] row-major == w_out rows, B-frag contiguous)
//   k2: per-head sequential scan, 1 block/head, 4 waves; A/B/C as f16
//       loop-invariant WMMA B-operand fragments in registers; state via LDS;
//       branch-free tanh (v_exp_f32 + v_rcp_f32); double-buffered x loads
//   k3: projection GEMM out = tokens + y @ w_out^T + b_out  (WMMA f16)
// ---------------------------------------------------------------------------

typedef __attribute__((ext_vector_type(16))) _Float16 v16h;
typedef __attribute__((ext_vector_type(8)))  float    v8f;

struct __align__(16) H8 { _Float16 h[8]; };
struct __align__(8)  H4 { _Float16 h[4]; };

// Branch-free tanh: 1 - 2/(exp2(2*log2e*x) + 1).
// Saturates correctly for large |x| (exp2->inf => 1; exp2->0 => -1).
static __device__ __forceinline__ float fast_tanh(float x) {
  const float e = __builtin_amdgcn_exp2f(x * 2.88539008177793f); // 2*log2(e)
  const float r = __builtin_amdgcn_rcpf(e + 1.0f);
  return fmaf(-2.0f, r, 1.0f);
}

// WMMA A-operand fragment (16x32 f16), ISA 7.12.2 layout.
// src is row-major [16+][ld] f16, 16-byte aligned rows.
static __device__ __forceinline__ v16h load_fragA(const _Float16* src, int ld,
                                                  int lane, int kbase) {
  const int m  = lane & 15;
  const int hi = (lane >> 4) & 1;           // lanes 16..31 take K+8 / K+24
  const H8 lo8 = *reinterpret_cast<const H8*>(src + m * ld + kbase + hi * 8);
  const H8 hi8 = *reinterpret_cast<const H8*>(src + m * ld + kbase + 16 + hi * 8);
  v16h v;
#pragma unroll
  for (int i = 0; i < 8; ++i) { v[i] = lo8.h[i]; v[8 + i] = hi8.h[i]; }
  return v;
}

// WMMA B-operand fragment (32x16 f16). srcT is [n][k] ("transposed") row-major
// so each lane's 16 K-values are two contiguous runs of 8 (at K and K+16).
static __device__ __forceinline__ v16h load_fragB(const _Float16* srcT, int ld,
                                                  int lane, int ncol0, int kbase) {
  const int n = ncol0 + (lane & 15);
  const int k = kbase + ((lane >> 4) & 1) * 8; // lanes 16..31 take K+8 / K+24
  const H8 lo8 = *reinterpret_cast<const H8*>(srcT + n * ld + k);
  const H8 hi8 = *reinterpret_cast<const H8*>(srcT + n * ld + k + 16);
  v16h v;
#pragma unroll
  for (int i = 0; i < 8; ++i) { v[i] = lo8.h[i]; v[8 + i] = hi8.h[i]; }
  return v;
}

static __device__ __forceinline__ v8f wmma_f16(v16h a, v16h b, v8f c) {
  return __builtin_amdgcn_wmma_f32_16x16x32_f16(false, a, false, b,
                                                (short)0, c, false, false);
}

// ---------------------------------------------------------------------------
// Kernel 0: RMSNorm, write f16 normalized x in [H][B][T][D] layout.
// One block per (b,t) token, 256 threads.
// ---------------------------------------------------------------------------
__global__ __launch_bounds__(256)
void k_rmsnorm(const float* __restrict__ tokens, const float* __restrict__ norm_w,
               _Float16* __restrict__ xn) {
  const int bt  = blockIdx.x;          // 0..B*T-1
  const int b   = bt >> 11;            // /2048
  const int t   = bt & 2047;
  const int tid = threadIdx.x;
  __shared__ float red[256];

  float ss = 0.f;
#pragma unroll
  for (int d = tid; d < 1024; d += 256) {
    float v = tokens[(size_t)bt * 1024 + d];
    ss += v * v;
  }
  red[tid] = ss;
  __syncthreads();
#pragma unroll
  for (int s = 128; s > 0; s >>= 1) {
    if (tid < s) red[tid] += red[tid + s];
    __syncthreads();
  }
  const float r = rsqrtf(red[0] * (1.0f / 1024.0f) + 1e-4f);

#pragma unroll
  for (int d = tid; d < 1024; d += 256) {
    float v = tokens[(size_t)bt * 1024 + d] * r * norm_w[d];
    const int h = d >> 6, dd = d & 63;
    xn[(((size_t)h * 8 + b) * 2048 + t) * 64 + dd] = (_Float16)v;
  }
}

// ---------------------------------------------------------------------------
// Kernel 1: w_out f32 -> f16 (layout preserved: [n][k] row-major).
// ---------------------------------------------------------------------------
__global__ __launch_bounds__(256)
void k_wcvt(const float* __restrict__ w, _Float16* __restrict__ wf) {
  const size_t i = (size_t)blockIdx.x * 256 + threadIdx.x;
  wf[i] = (_Float16)w[i];
}

// ---------------------------------------------------------------------------
// Kernel 2: sequential scan. 1 block per head (grid=16), 128 threads (4 waves).
// Wave w owns output columns [16w, 16w+16).
// state tile is 16x64 (rows 0..7 = batch, rows 8..15 stay zero).
// ---------------------------------------------------------------------------
__global__ __launch_bounds__(128)
void k_scan(const _Float16* __restrict__ xn, const float* __restrict__ a,
            const float* __restrict__ b, const float* __restrict__ c,
            const float* __restrict__ skip, _Float16* __restrict__ y) {
  const int h    = blockIdx.x;
  const int tid  = threadIdx.x;
  const int lane = tid & 31;
  const int wv   = tid >> 5;
  const int ncol0 = wv * 16;

  __shared__ __align__(16) _Float16 ATl[64 * 64]; // tanh(a)^T : [t'][s]
  __shared__ __align__(16) _Float16 BTl[64 * 64]; // b^T       : [s][d]
  __shared__ __align__(16) _Float16 CTl[64 * 64]; // c^T       : [d][s]
  __shared__ __align__(16) _Float16 st[16 * 64];  // state f16 (rows 8..15 zero)
  __shared__ __align__(16) _Float16 xt[16 * 64];  // x_t  f16 (rows 8..15 zero)

  // Preload transposed f16 copies of tanh(a[h]), b[h], c[h].
  for (int idx = tid; idx < 4096; idx += 128) {
    const int n = idx >> 6, k = idx & 63;
    ATl[idx] = (_Float16)fast_tanh(a[((size_t)h * 64 + k) * 64 + n]); // a_t[k][n]
    BTl[idx] = (_Float16)b[((size_t)h * 64 + k) * 64 + n];            // b[d=k][s=n]
    CTl[idx] = (_Float16)c[((size_t)h * 64 + k) * 64 + n];            // c[s=k][d=n]
  }
  for (int idx = tid; idx < 1024; idx += 128) {
    st[idx] = (_Float16)0.f;
    xt[idx] = (_Float16)0.f;
  }
  __syncthreads();

  // Loop-invariant B-operand fragments (live in registers across all T steps).
  v16h bA0 = load_fragB(ATl, 64, lane, ncol0, 0);
  v16h bA1 = load_fragB(ATl, 64, lane, ncol0, 32);
  v16h bB0 = load_fragB(BTl, 64, lane, ncol0, 0);
  v16h bB1 = load_fragB(BTl, 64, lane, ncol0, 32);
  v16h bC0 = load_fragB(CTl, 64, lane, ncol0, 0);
  v16h bC1 = load_fragB(CTl, 64, lane, ncol0, 32);

  const int ncol = ncol0 + (lane & 15);
  const int mb   = ((lane >> 4) & 1) ? 8 : 0; // C/D row base for this lane
  const float skv = skip[h * 64 + ncol];

  // Double-buffered x: each thread owns 4 contiguous halves of the 8x64 tile.
  const int idx = tid * 4;
  const int xm = idx >> 6, xd = idx & 63;
  const _Float16* gx = xn + (((size_t)h * 8 + xm) * 2048) * 64 + xd;
  H4 xreg = *reinterpret_cast<const H4*>(gx); // t = 0

  for (int t = 0; t < 2048; ++t) {
    __syncthreads(); // previous step's readers of xt are done
    *reinterpret_cast<H4*>(&xt[idx]) = xreg;
    // Issue next step's global load now; latency hides under this step's math.
    const int tn = (t + 1 < 2048) ? t + 1 : t;
    const H4 xnext = *reinterpret_cast<const H4*>(gx + (size_t)tn * 64);
    __syncthreads();

    // pre = state @ tanh(a) + x @ b      (4 x v_wmma_f32_16x16x32_f16)
    v16h sa0 = load_fragA(st, 64, lane, 0);
    v16h sa1 = load_fragA(st, 64, lane, 32);
    v16h xa0 = load_fragA(xt, 64, lane, 0);
    v16h xa1 = load_fragA(xt, 64, lane, 32);
    v8f acc = {};
    acc = wmma_f16(sa0, bA0, acc);
    acc = wmma_f16(sa1, bA1, acc);
    acc = wmma_f16(xa0, bB0, acc);
    acc = wmma_f16(xa1, bB1, acc);

    float ns[8];
#pragma unroll
    for (int r = 0; r < 8; ++r) ns[r] = fast_tanh(acc[r]);

    __syncthreads(); // everyone finished reading old state
#pragma unroll
    for (int r = 0; r < 8; ++r)
      st[(mb + r) * 64 + ncol] = (_Float16)ns[r];
    __syncthreads(); // new state visible to all waves

    // mixed = state @ c                  (2 x v_wmma_f32_16x16x32_f16)
    v16h na0 = load_fragA(st, 64, lane, 0);
    v16h na1 = load_fragA(st, 64, lane, 32);
    v8f acc2 = {};
    acc2 = wmma_f16(na0, bC0, acc2);
    acc2 = wmma_f16(na1, bC1, acc2);

    if (mb == 0) { // only rows 0..7 are real batch rows
#pragma unroll
      for (int r = 0; r < 8; ++r) {
        const float xv = (float)xt[r * 64 + ncol];
        const float o  = acc2[r] + skv * xv;
        y[(((size_t)r * 2048 + t) * 1024) + h * 64 + ncol] = (_Float16)o;
      }
    }
    xreg = xnext;
  }
}

// ---------------------------------------------------------------------------
// Kernel 3: out = tokens + y @ w_out^T + b_out.
// Block = 128 threads (4 waves); wave w does n-tile blockIdx.x*64 + 16w,
// 16 rows per block (blockIdx.y). K=1024 in 32 steps of k=32.
// ---------------------------------------------------------------------------
__global__ __launch_bounds__(128)
void k_proj(const float* __restrict__ tokens, const float* __restrict__ b_out,
            const _Float16* __restrict__ y, const _Float16* __restrict__ wf,
            float* __restrict__ out) {
  const int tid  = threadIdx.x;
  const int lane = tid & 31;
  const int wv   = tid >> 5;
  const int row0  = blockIdx.y * 16;
  const int ncol0 = blockIdx.x * 64 + wv * 16;

  v8f acc = {};
  const _Float16* arow = y + (size_t)row0 * 1024;
#pragma unroll 4
  for (int kb = 0; kb < 1024; kb += 32) {
    v16h af = load_fragA(arow, 1024, lane, kb);
    v16h bf = load_fragB(wf, 1024, lane, ncol0, kb); // wf[n][k]: w_out row n
    acc = wmma_f16(af, bf, acc);
  }

  const int n  = ncol0 + (lane & 15);
  const int mb = ((lane >> 4) & 1) ? 8 : 0;
  const float bo = b_out[n];
#pragma unroll
  for (int r = 0; r < 8; ++r) {
    const size_t m = (size_t)(row0 + mb + r);
    out[m * 1024 + n] = tokens[m * 1024 + n] + acc[r] + bo;
  }
}

// ---------------------------------------------------------------------------
extern "C" void kernel_launch(void* const* d_in, const int* in_sizes, int n_in,
                              void* d_out, int out_size, void* d_ws, size_t ws_size,
                              hipStream_t stream) {
  const float* tokens = (const float*)d_in[0];
  const float* norm_w = (const float*)d_in[1];
  const float* a      = (const float*)d_in[2];
  const float* b      = (const float*)d_in[3];
  const float* c      = (const float*)d_in[4];
  const float* skip   = (const float*)d_in[5];
  const float* w_out  = (const float*)d_in[6];
  const float* b_out  = (const float*)d_in[7];
  float* out = (float*)d_out;

  // Workspace carve-up (f16): xn 32MB, y 32MB, wf 2MB.
  _Float16* xn = (_Float16*)d_ws;
  _Float16* yb = xn + (size_t)8 * 2048 * 1024;
  _Float16* wf = yb + (size_t)8 * 2048 * 1024;

  k_rmsnorm<<<8 * 2048, 256, 0, stream>>>(tokens, norm_w, xn);
  k_wcvt<<<(1024 * 1024) / 256, 256, 0, stream>>>(w_out, wf);
  k_scan<<<16, 128, 0, stream>>>(xn, a, b, c, skip, yb);
  k_proj<<<dim3(16, 1024), 128, 0, stream>>>(tokens, b_out, yb, wf, out);
}
